// SoftmaxFreeAttentionKernel_49838800503102
// MI455X (gfx1250) — compile-verified, matrix operand order checked by reading, not run on hardware
//
#include <hip/hip_runtime.h>
#include <hip/hip_bf16.h>

#define DEVINL __device__ __forceinline__

constexpr int B_ = 2, H_ = 16, N_ = 4096, D_ = 64, M_ = 128, RATIO_ = 32;
constexpr int BH_ = B_ * H_;
constexpr float SCALE = 0.35355339059327373f;   // 64^-0.25

// ---------------- workspace layout (bytes) ----------------
constexpr size_t OFF_QH  = 0;                                   // f16 [BH,N,D]
constexpr size_t OFF_VH  = OFF_QH  + (size_t)BH_*N_*D_*2;       // f16 [BH,N,D]
constexpr size_t OFF_QS  = OFF_VH  + (size_t)BH_*N_*D_*2;       // f32 [BH,N]
constexpr size_t OFF_QLF = OFF_QS  + (size_t)BH_*N_*4;          // f32 [BH,M,D]
constexpr size_t OFF_QLH = OFF_QLF + (size_t)BH_*M_*D_*4;       // f16 [BH,M,D]
constexpr size_t OFF_QLS = OFF_QLH + (size_t)BH_*M_*D_*2;       // f32 [BH,M]
constexpr size_t OFF_KV  = OFF_QLS + (size_t)BH_*M_*4;          // f32 [BH,M,D]
constexpr size_t OFF_WH  = OFF_KV  + (size_t)BH_*M_*D_*4;       // f16 [BH,M,D]

// ---------------- WMMA types / helpers ----------------
typedef __attribute__((ext_vector_type(2)))  float    v2f;
typedef __attribute__((ext_vector_type(8)))  float    v8f;
typedef __attribute__((ext_vector_type(16))) _Float16 v16h;

struct h2 { _Float16 x, y; };
union V16U { v16h v; uint4 q[2]; };

DEVINL v8f v8zero() { v8f z = {0.f,0.f,0.f,0.f,0.f,0.f,0.f,0.f}; return z; }

// CDNA5 async global->LDS copy (16B per lane), tracked by ASYNCcnt.
// vdst = workgroup-relative LDS byte offset; addr = 64-bit global address.
DEVINL void async_ld_b128(unsigned lds_off, const void* gptr) {
  asm volatile("global_load_async_to_lds_b128 %0, %1, off"
               :: "v"(lds_off), "v"(gptr) : "memory");
}
DEVINL void async_wait0() {
  asm volatile("s_wait_asynccnt 0" ::: "memory");
}
// low 32 bits of a flat pointer into the LDS aperture == LDS byte offset
DEVINL unsigned lds_off(const void* p) { return (unsigned)(uintptr_t)p; }

// 16-bit A/B fragment from a row-major tile row (matching ISA 16-bit A layout:
// lanes 0-15 hold K {0..7,16..23}, lanes 16-31 hold K {8..15,24..31})
DEVINL v16h hfrag_row(const _Float16* rowk0, int lane) {
  int klo = (lane & 16) ? 8 : 0;
  V16U u;
  u.q[0] = *(const uint4*)(rowk0 + klo);
  u.q[1] = *(const uint4*)(rowk0 + 16 + klo);
  return u.v;
}
// same fragment but gathering a column of a row-major matrix (transposed operand)
DEVINL v16h hfrag_col(const _Float16* base, int ld, int k0, int col, int lane) {
  int klo = (lane & 16) ? 8 : 0;
  v16h v;
#pragma unroll
  for (int h = 0; h < 8; ++h) v[h]     = base[(k0 + klo + h) * ld + col];
#pragma unroll
  for (int h = 0; h < 8; ++h) v[h + 8] = base[(k0 + 16 + klo + h) * ld + col];
  return v;
}
DEVINL v8f wmma_f16(v16h a, v16h b, v8f c) {
  return __builtin_amdgcn_wmma_f32_16x16x32_f16(false, a, false, b, (short)0, c, false, false);
}
// f32 WMMA 16x16x4: A 16x4 (2 VGPR: lanes0-15 K{0,1}, lanes16-31 K{2,3})
DEVINL v2f afrag32(const float* A, int ld, int r0, int k0, int lane) {
  int klo = (lane & 16) ? 2 : 0;
  const float* p = A + (size_t)(r0 + (lane & 15)) * ld + k0 + klo;
  v2f a; a.x = p[0]; a.y = p[1]; return a;
}
DEVINL v2f bfrag32(const float* Bm, int ld, int k0, int c0, int lane) {
  int klo = (lane & 16) ? 2 : 0;
  const float* p = Bm + (size_t)(k0 + klo) * ld + c0 + (lane & 15);
  v2f b; b.x = p[0]; b.y = p[ld]; return b;
}
// f32 WMMA allows no A/B negate (NEG must be 0) -> all modifiers constant zero
DEVINL v8f wmma_f32(v2f a, v2f b, v8f c) {
  return __builtin_amdgcn_wmma_f32_16x16x4_f32(false, a, false, b, (short)0, c, false, false);
}

// ---------------- kernel 1: Qh=f16(Q*scale), Vh=f16(V), qs=||Qs||^2 ----------------
__global__ __launch_bounds__(256)
void prep_kernel(const float* __restrict__ Q, const float* __restrict__ V,
                 _Float16* __restrict__ Qh, _Float16* __restrict__ Vh,
                 float* __restrict__ qs) {
  int lane = threadIdx.x & 31;
  int warp = threadIdx.x >> 5;
  size_t row = (size_t)blockIdx.x * 8 + warp;   // < BH*N
  float2 qa = ((const float2*)(Q + row * D_))[lane];
  float2 va = ((const float2*)(V + row * D_))[lane];
  float s0 = qa.x * SCALE, s1 = qa.y * SCALE;
  ((h2*)(Qh + row * D_))[lane] = h2{(_Float16)s0, (_Float16)s1};
  ((h2*)(Vh + row * D_))[lane] = h2{(_Float16)va.x, (_Float16)va.y};
  float ss = s0 * s0 + s1 * s1;
#pragma unroll
  for (int off = 16; off; off >>= 1) ss += __shfl_xor(ss, off);
  if (lane == 0) qs[row] = ss;
}

// ---------------- kernel 2: landmarks (conv1d stride=kernel=32 + LN + GELU) ----------------
__global__ __launch_bounds__(64)
void landmark_kernel(const float* __restrict__ Q, const float* __restrict__ W,
                     const float* __restrict__ gamma, const float* __restrict__ beta,
                     float* __restrict__ QLf, _Float16* __restrict__ QLh,
                     float* __restrict__ qls) {
  __shared__ float chunk[RATIO_ * D_];   // [k][din]
  __shared__ float sred[D_];
  int bhm = blockIdx.x;                  // bh*M + m ; rows (bh*N + m*32) == bhm*32 (contiguous)
  const float* src = Q + (size_t)bhm * RATIO_ * D_;
  for (int i = threadIdx.x; i < RATIO_ * D_; i += 64) chunk[i] = src[i];
  __syncthreads();
  int dout = threadIdx.x;
  const float* wr = W + (size_t)dout * D_ * RATIO_;
  float acc = 0.f;
  for (int din = 0; din < D_; ++din) {
    const float* wd = wr + din * RATIO_;
#pragma unroll 8
    for (int k = 0; k < RATIO_; ++k) acc += chunk[k * D_ + din] * wd[k];
  }
  acc *= SCALE;                                  // conv is linear in Qs = Q*scale
  sred[dout] = acc; __syncthreads();
  float mu = 0.f;
  for (int i = 0; i < D_; ++i) mu += sred[i];
  mu *= (1.f / D_);
  float var = 0.f;
  for (int i = 0; i < D_; ++i) { float dx = sred[i] - mu; var += dx * dx; }
  var *= (1.f / D_);
  float y = (acc - mu) * rsqrtf(var + 1e-5f) * gamma[dout] + beta[dout];
  float g = 0.5f * y * (1.f + erff(y * 0.70710678118654752f));   // exact GELU
  QLf[(size_t)bhm * D_ + dout] = g;
  QLh[(size_t)bhm * D_ + dout] = (_Float16)g;
  __syncthreads();
  sred[dout] = g * g; __syncthreads();
  if (dout == 0) { float s = 0.f; for (int i = 0; i < D_; ++i) s += sred[i]; qls[bhm] = s; }
}

__global__ void zero_f32(float* p, int n) {
  int i = blockIdx.x * blockDim.x + threadIdx.x;
  if (i < n) p[i] = 0.f;
}

// ---------------- kernel 3: kv += k1^T @ V (k1 recomputed per tile) ----------------
__global__ __launch_bounds__(256)
void passA_kernel(const _Float16* __restrict__ Qh, const _Float16* __restrict__ Vh,
                  const float* __restrict__ qs, const _Float16* __restrict__ QLh,
                  const float* __restrict__ qls, float* __restrict__ kv) {
  __shared__ _Float16 sQ[128 * 64];
  __shared__ _Float16 sV[128 * 64];
  __shared__ _Float16 sQL[128 * 64];
  __shared__ _Float16 sK1[128 * 128];
  __shared__ float sqn[128];
  __shared__ float sqm[128];
  int bh = blockIdx.x >> 5, c = blockIdx.x & 31;
  size_t n0row = (size_t)bh * N_ + (size_t)c * 128;
  {
    // async global->LDS staging (16B per lane per issue), ASYNCcnt-tracked
    const _Float16* gq = Qh + n0row * D_;
    const _Float16* gv = Vh + n0row * D_;
    const _Float16* gl = QLh + (size_t)bh * M_ * D_;
#pragma unroll
    for (int i = 0; i < 4; ++i) {
      int e = (threadIdx.x + 256 * i) * 8;       // element index, 8 halfs = 16B
      async_ld_b128(lds_off(&sQ[e]),  gq + e);
      async_ld_b128(lds_off(&sV[e]),  gv + e);
      async_ld_b128(lds_off(&sQL[e]), gl + e);
    }
    if (threadIdx.x < 128) {
      sqn[threadIdx.x] = qs[n0row + threadIdx.x];
      sqm[threadIdx.x] = qls[(size_t)bh * M_ + threadIdx.x];
    }
    async_wait0();
  }
  __syncthreads();
  int lane = threadIdx.x & 31, w = threadIdx.x >> 5, hi8 = (lane & 16) ? 8 : 0;
  // S = Qs . QL^T  (wave w owns rows [16w,16w+16), all 8 col tiles; K=64)
  v8f acc[8];
#pragma unroll
  for (int j = 0; j < 8; ++j) acc[j] = v8zero();
#pragma unroll
  for (int kk = 0; kk < 64; kk += 32) {
    v16h a = hfrag_row(&sQ[(16 * w + (lane & 15)) * 64 + kk], lane);
#pragma unroll
    for (int j = 0; j < 8; ++j) {
      v16h b = hfrag_row(&sQL[(16 * j + (lane & 15)) * 64 + kk], lane);
      acc[j] = wmma_f16(a, b, acc[j]);
    }
  }
  // k1 = exp(-0.5*max(qn+qm-2S,0))  -> LDS (f16)
#pragma unroll
  for (int j = 0; j < 8; ++j)
#pragma unroll
    for (int r = 0; r < 8; ++r) {
      int row = 16 * w + r + hi8, col = 16 * j + (lane & 15);
      float d = fmaxf(sqn[row] + sqm[col] - 2.f * acc[j][r], 0.f);
      sK1[row * 128 + col] = (_Float16)__expf(-0.5f * d);
    }
  __syncthreads();
  // kv_tile = k1^T @ V  (wave w owns m-rows [16w,16w+16), 4 d tiles; K=128)
  v8f a2[4];
#pragma unroll
  for (int j = 0; j < 4; ++j) a2[j] = v8zero();
  for (int k0 = 0; k0 < 128; k0 += 32) {
    v16h a = hfrag_col(sK1, 128, k0, 16 * w + (lane & 15), lane);
#pragma unroll
    for (int j = 0; j < 4; ++j) {
      v16h b = hfrag_col(sV, 64, k0, 16 * j + (lane & 15), lane);
      a2[j] = wmma_f16(a, b, a2[j]);
    }
  }
  float* kvb = kv + (size_t)bh * M_ * D_;
#pragma unroll
  for (int j = 0; j < 4; ++j)
#pragma unroll
    for (int r = 0; r < 8; ++r) {
      int m = 16 * w + r + hi8, d = 16 * j + (lane & 15);
      atomicAdd(&kvb[m * 64 + d], a2[j][r]);
    }
}

// ---------------- kernel 4: k2, Newton-Schulz inverse, W = k2inv @ kv ----------------
__global__ __launch_bounds__(256)
void k2inv_kernel(const float* __restrict__ QLf, const float* __restrict__ qls,
                  const float* __restrict__ kv, _Float16* __restrict__ Wh) {
  __shared__ float sP[128 * 128];
  __shared__ float sVm[128 * 128];
  __shared__ float sT[128 * 128];
  __shared__ float sred[128];
  __shared__ float salpha;
  __shared__ unsigned smax;
  __shared__ int sdone;
  int bh = blockIdx.x, tid = threadIdx.x;
  int lane = tid & 31, w = tid >> 5, hi8 = (lane & 16) ? 8 : 0;
  // stage QL (f32) into sT
  const float* gql = QLf + (size_t)bh * M_ * D_;
  for (int i = tid; i < 128 * 64; i += 256) sT[i] = gql[i];
  if (tid < 128) sred[tid] = qls[(size_t)bh * M_ + tid];
  if (tid == 0) { smax = 0u; sdone = 0; salpha = 0.f; }
  __syncthreads();
  // P = k2 = exp(-0.5*max(||a||+||b||-2ab,0))
  for (int idx = tid; idx < 128 * 128; idx += 256) {
    int i = idx >> 7, j = idx & 127;
    const float* a = &sT[i * 64]; const float* b = &sT[j * 64];
    float dot = 0.f;
#pragma unroll 8
    for (int d = 0; d < 64; ++d) dot += a[d] * b[d];
    float dd = fmaxf(sred[i] + sred[j] - 2.f * dot, 0.f);
    sP[idx] = __expf(-0.5f * dd);
  }
  __syncthreads();
  // alpha = 2 / max_rowsum(P)^2
  if (tid < 128) {
    float rs = 0.f;
    for (int j = 0; j < 128; ++j) rs += sP[tid * 128 + j];
    sred[tid] = rs;
  }
  __syncthreads();
  if (tid == 0) {
    float mx = 0.f;
    for (int i = 0; i < 128; ++i) mx = fmaxf(mx, sred[i]);
    salpha = 2.f / (mx * mx);
  }
  __syncthreads();
  // P2 = P @ P  -> sT   (f32 WMMA)
  {
    v8f acc[8];
#pragma unroll
    for (int j = 0; j < 8; ++j) acc[j] = v8zero();
    for (int k0 = 0; k0 < 128; k0 += 4) {
      v2f a = afrag32(sP, 128, 16 * w, k0, lane);
#pragma unroll
      for (int j = 0; j < 8; ++j) acc[j] = wmma_f32(a, bfrag32(sP, 128, k0, 16 * j, lane), acc[j]);
    }
#pragma unroll
    for (int j = 0; j < 8; ++j)
#pragma unroll
      for (int r = 0; r < 8; ++r)
        sT[(16 * w + r + hi8) * 128 + 16 * j + (lane & 15)] = acc[j][r];
  }
  __syncthreads();
  // 10 alpha-halving trials using induced 1-norm of (I - alpha*P2)
#pragma unroll 1
  for (int t = 0; t < 10; ++t) {
    float alpha = salpha;
    if (tid < 128) {
      float cs = 0.f;
      for (int i = 0; i < 128; ++i)
        cs += fabsf(((i == tid) ? 1.f : 0.f) - alpha * sT[i * 128 + tid]);
      sred[tid] = cs;
    }
    __syncthreads();
    if (tid == 0) {
      float mx = 0.f;
      for (int i = 0; i < 128; ++i) mx = fmaxf(mx, sred[i]);
      if (mx > 1.01f) salpha = alpha * 0.5f;
    }
    __syncthreads();
  }
  // Vm = alpha * P
  {
    float alpha = salpha;
    for (int i = tid; i < 128 * 128; i += 256) sVm[i] = alpha * sP[i];
  }
  __syncthreads();
  // Newton loop: C = 2Vm - Vm@P@Vm  (skip once diverged, like reference)
#pragma unroll 1
  for (int it = 0; it < 20; ++it) {
    if (sdone) break;
    // T = P @ Vm
    {
      v8f acc[8];
#pragma unroll
      for (int j = 0; j < 8; ++j) acc[j] = v8zero();
      for (int k0 = 0; k0 < 128; k0 += 4) {
        v2f a = afrag32(sP, 128, 16 * w, k0, lane);
#pragma unroll
        for (int j = 0; j < 8; ++j) acc[j] = wmma_f32(a, bfrag32(sVm, 128, k0, 16 * j, lane), acc[j]);
      }
      __syncthreads();   // everyone done reading old sT
#pragma unroll
      for (int j = 0; j < 8; ++j)
#pragma unroll
        for (int r = 0; r < 8; ++r)
          sT[(16 * w + r + hi8) * 128 + 16 * j + (lane & 15)] = acc[j][r];
    }
    __syncthreads();
    // want C = 2*Vm - Vm@T ; f32 WMMA has no A/B negate, so compute
    // acc = (-2*Vm) + Vm@T  and negate at the end: C = -acc.
    v8f acc[8];
#pragma unroll
    for (int j = 0; j < 8; ++j)
#pragma unroll
      for (int r = 0; r < 8; ++r)
        acc[j][r] = -2.f * sVm[(16 * w + r + hi8) * 128 + 16 * j + (lane & 15)];
    for (int k0 = 0; k0 < 128; k0 += 4) {
      v2f a = afrag32(sVm, 128, 16 * w, k0, lane);
#pragma unroll
      for (int j = 0; j < 8; ++j) acc[j] = wmma_f32(a, bfrag32(sT, 128, k0, 16 * j, lane), acc[j]);
    }
    float lm = 0.f;
#pragma unroll
    for (int j = 0; j < 8; ++j)
#pragma unroll
      for (int r = 0; r < 8; ++r) lm = fmaxf(lm, fabsf(acc[j][r]));
    atomicMax(&smax, __float_as_uint(lm));   // non-negative floats: bit order == value order
    __syncthreads();
    bool bad = __uint_as_float(smax) > 10000.f;
    if (!bad) {
#pragma unroll
      for (int j = 0; j < 8; ++j)
#pragma unroll
        for (int r = 0; r < 8; ++r)
          sVm[(16 * w + r + hi8) * 128 + 16 * j + (lane & 15)] = -acc[j][r];
    }
    if (tid == 0) { if (bad) sdone = 1; smax = 0u; }
    __syncthreads();
  }
  // W = Vm @ kv  (128x64, K=128) -> f16
  {
    float* sKV = sT;
    const float* gkv = kv + (size_t)bh * M_ * D_;
    for (int i = tid; i < 128 * 64; i += 256) sKV[i] = gkv[i];
    __syncthreads();
    v8f acc[4];
#pragma unroll
    for (int j = 0; j < 4; ++j) acc[j] = v8zero();
    for (int k0 = 0; k0 < 128; k0 += 4) {
      v2f a = afrag32(sVm, 128, 16 * w, k0, lane);
#pragma unroll
      for (int j = 0; j < 4; ++j) acc[j] = wmma_f32(a, bfrag32(sKV, 64, k0, 16 * j, lane), acc[j]);
    }
    _Float16* gw = Wh + (size_t)bh * M_ * D_;
#pragma unroll
    for (int j = 0; j < 4; ++j)
#pragma unroll
      for (int r = 0; r < 8; ++r)
        gw[(16 * w + r + hi8) * 64 + 16 * j + (lane & 15)] = (_Float16)acc[j][r];
  }
}

// ---------------- kernel 5: X = k1 @ W + depthwise conv(V) ----------------
__global__ __launch_bounds__(256)
void passB_kernel(const _Float16* __restrict__ Qh, const float* __restrict__ qs,
                  const _Float16* __restrict__ QLh, const float* __restrict__ qls,
                  const _Float16* __restrict__ Wh, const float* __restrict__ V,
                  const float* __restrict__ convw, float* __restrict__ out) {
  __shared__ _Float16 sQ[128 * 64];
  __shared__ _Float16 sQL[128 * 64];
  __shared__ _Float16 sW[128 * 64];
  __shared__ _Float16 sK1[128 * 128];
  __shared__ float sqn[128];
  __shared__ float sqm[128];
  int bh = blockIdx.x >> 5, c = blockIdx.x & 31;
  size_t n0row = (size_t)bh * N_ + (size_t)c * 128;
  {
    const _Float16* gq = Qh + n0row * D_;
    const _Float16* gl = QLh + (size_t)bh * M_ * D_;
    const _Float16* gw = Wh + (size_t)bh * M_ * D_;
#pragma unroll
    for (int i = 0; i < 4; ++i) {
      int e = (threadIdx.x + 256 * i) * 8;
      async_ld_b128(lds_off(&sQ[e]),  gq + e);
      async_ld_b128(lds_off(&sQL[e]), gl + e);
      async_ld_b128(lds_off(&sW[e]),  gw + e);
    }
    if (threadIdx.x < 128) {
      sqn[threadIdx.x] = qs[n0row + threadIdx.x];
      sqm[threadIdx.x] = qls[(size_t)bh * M_ + threadIdx.x];
    }
    async_wait0();
  }
  __syncthreads();
  int lane = threadIdx.x & 31, w = threadIdx.x >> 5, hi8 = (lane & 16) ? 8 : 0;
  // S and k1 tile (same as passA)
  v8f acc[8];
#pragma unroll
  for (int j = 0; j < 8; ++j) acc[j] = v8zero();
#pragma unroll
  for (int kk = 0; kk < 64; kk += 32) {
    v16h a = hfrag_row(&sQ[(16 * w + (lane & 15)) * 64 + kk], lane);
#pragma unroll
    for (int j = 0; j < 8; ++j) {
      v16h b = hfrag_row(&sQL[(16 * j + (lane & 15)) * 64 + kk], lane);
      acc[j] = wmma_f16(a, b, acc[j]);
    }
  }
#pragma unroll
  for (int j = 0; j < 8; ++j)
#pragma unroll
    for (int r = 0; r < 8; ++r) {
      int row = 16 * w + r + hi8, col = 16 * j + (lane & 15);
      float d = fmaxf(sqn[row] + sqm[col] - 2.f * acc[j][r], 0.f);
      sK1[row * 128 + col] = (_Float16)__expf(-0.5f * d);
    }
  __syncthreads();
  // X = k1 @ W   (128x64, K=128)
  v8f xo[4];
#pragma unroll
  for (int j = 0; j < 4; ++j) xo[j] = v8zero();
  for (int k0 = 0; k0 < 128; k0 += 32) {
    v16h a = hfrag_row(&sK1[(16 * w + (lane & 15)) * 128 + k0], lane);
#pragma unroll
    for (int j = 0; j < 4; ++j) {
      v16h b = hfrag_col(sW, 64, k0, 16 * j + (lane & 15), lane);
      xo[j] = wmma_f16(a, b, xo[j]);
    }
  }
  // depthwise conv over sequence + store
  int h = bh & (H_ - 1);
  float w0 = convw[h * 3 + 0], w1 = convw[h * 3 + 1], w2 = convw[h * 3 + 2];
  const float* Vb = V + (size_t)bh * N_ * D_;
  float* ob = out + (size_t)bh * N_ * D_;
#pragma unroll
  for (int j = 0; j < 4; ++j)
#pragma unroll
    for (int r = 0; r < 8; ++r) {
      int n = c * 128 + 16 * w + r + hi8;
      int d = 16 * j + (lane & 15);
      const float* vp = Vb + (size_t)n * D_ + d;
      float xc = w1 * vp[0];
      if (n > 0)      xc += w0 * vp[-D_];
      if (n < N_ - 1) xc += w2 * vp[D_];
      ob[(size_t)n * D_ + d] = xo[j][r] + xc;
    }
}

// ---------------- launcher ----------------
extern "C" void kernel_launch(void* const* d_in, const int* in_sizes, int n_in,
                              void* d_out, int out_size, void* d_ws, size_t ws_size,
                              hipStream_t stream) {
  const float* Q     = (const float*)d_in[0];
  const float* V     = (const float*)d_in[1];
  const float* Qw    = (const float*)d_in[2];
  const float* gamma = (const float*)d_in[3];
  const float* beta  = (const float*)d_in[4];
  const float* convw = (const float*)d_in[5];
  float* out = (float*)d_out;
  char* ws = (char*)d_ws;
  _Float16* Qh  = (_Float16*)(ws + OFF_QH);
  _Float16* Vh  = (_Float16*)(ws + OFF_VH);
  float*    qsq = (float*)(ws + OFF_QS);
  float*    QLf = (float*)(ws + OFF_QLF);
  _Float16* QLh = (_Float16*)(ws + OFF_QLH);
  float*    qls = (float*)(ws + OFF_QLS);
  float*    kv  = (float*)(ws + OFF_KV);
  _Float16* Wh  = (_Float16*)(ws + OFF_WH);

  prep_kernel<<<BH_ * N_ / 8, 256, 0, stream>>>(Q, V, Qh, Vh, qsq);
  landmark_kernel<<<BH_ * M_, 64, 0, stream>>>(Q, Qw, gamma, beta, QLf, QLh, qls);
  zero_f32<<<(BH_ * M_ * D_ + 255) / 256, 256, 0, stream>>>(kv, BH_ * M_ * D_);
  passA_kernel<<<BH_ * (N_ / 128), 256, 0, stream>>>(Qh, Vh, qsq, QLh, qls, kv);
  k2inv_kernel<<<BH_, 256, 0, stream>>>(QLf, qls, kv, Wh);
  passB_kernel<<<BH_ * (N_ / 128), 256, 0, stream>>>(Qh, qsq, QLh, qls, Wh, V, convw, out);
}
